// GraphAttentionConv_60962765799609
// MI455X (gfx1250) — matrix operand order repaired and checked by reading, not compiled, not run on hardware
//
#include <hip/hip_runtime.h>

typedef __attribute__((ext_vector_type(16))) _Float16 v16h;
typedef __attribute__((ext_vector_type(8)))  _Float16 v8h;
typedef __attribute__((ext_vector_type(8)))  float    v8f;
typedef __attribute__((ext_vector_type(4)))  int      v4i;

#define NNODES 8192
#define INF_   256
#define OUTF   128
#define NCOLS  144   // 128 numerator cols + 1 denominator col + 15 zero pad
#define LDSP   40    // LDS pitch in halves (padded from 32 -> conflict-free)

#define AS1 __attribute__((address_space(1)))
#define AS3 __attribute__((address_space(3)))

#if defined(__has_builtin)
#if __has_builtin(__builtin_amdgcn_global_load_async_to_lds_b128) && \
    __has_builtin(__builtin_amdgcn_s_wait_asynccnt)
#define USE_ASYNC_LDS 1
#endif
#endif

// ---- order-preserving float<->uint encoding for atomicMax over floats ----
__device__ __forceinline__ unsigned enc_f32(float x) {
  unsigned u = __float_as_uint(x);
  return (u & 0x80000000u) ? ~u : (u | 0x80000000u);
}
__device__ __forceinline__ float dec_f32(unsigned k) {
  unsigned u = (k & 0x80000000u) ? (k & 0x7fffffffu) : ~k;
  return __uint_as_float(u);
}

// convert 8 contiguous f32 -> slots [base..base+7] of a v16h WMMA operand
__device__ __forceinline__ void cvt8(v16h& v, int base, const float* __restrict__ p) {
  const float4* q = (const float4*)p;
  float4 a = q[0], b = q[1];
  v[base+0]=(_Float16)a.x; v[base+1]=(_Float16)a.y; v[base+2]=(_Float16)a.z; v[base+3]=(_Float16)a.w;
  v[base+4]=(_Float16)b.x; v[base+5]=(_Float16)b.y; v[base+6]=(_Float16)b.z; v[base+7]=(_Float16)b.w;
}

// 8 adjacency int32 -> f16 0/1 mask (diagonal handled in the epilogue)
__device__ __forceinline__ void mask8(v16h& v, int base, const int* __restrict__ p) {
  const int4* q4 = (const int4*)p;
  int4 a = q4[0], b = q4[1];
  int vals[8] = {a.x,a.y,a.z,a.w,b.x,b.y,b.z,b.w};
#pragma unroll
  for (int e = 0; e < 8; ++e)
    v[base+e] = (vals[e] > 0) ? (_Float16)1.0f : (_Float16)0.0f;
}

// 8 contiguous f16 from LDS -> slots [base..base+7]
__device__ __forceinline__ void lds8(v16h& v, int base, const _Float16* p) {
  v8h q = *(const v8h*)p;
#pragma unroll
  for (int e = 0; e < 8; ++e) v[base+e] = q[e];
}

// ---------------- K0: init the global-max cell ----------------
__global__ void gat_k0_init(unsigned* maxcell) { *maxcell = 0u; }

// ---------------- K1: X' = X @ W^T + b  (WMMA f16 -> f32) ----------------
// block = 256 thr = 8 waves, each wave: 16 rows x 128 cols, K-loop over 256.
__global__ __launch_bounds__(256) void gat_k1_linear(
    const float* __restrict__ X, const float* __restrict__ W,
    const float* __restrict__ bias, float* __restrict__ Xp) {
  const int tid = threadIdx.x, wave = tid >> 5, lane = tid & 31;
  const int ln = lane & 15, grp = lane >> 4;
  const int i0 = blockIdx.x * 128;
  const int arow = i0 + wave * 16 + ln;          // A-matrix row for this lane
  v8f zero = {};
  v8f acc[8];
#pragma unroll
  for (int t = 0; t < 8; ++t) acc[t] = zero;

  for (int k0 = 0; k0 < INF_; k0 += 32) {
    v16h a;
    const float* xa = X + (size_t)arow * INF_ + k0 + grp * 8;
    cvt8(a, 0, xa); cvt8(a, 8, xa + 16);         // K {0-7,16-23} / {8-15,24-31}
#pragma unroll
    for (int t = 0; t < 8; ++t) {
      const float* wb = W + (size_t)(16 * t + ln) * INF_ + k0 + grp * 8;
      v16h b; cvt8(b, 0, wb); cvt8(b, 8, wb + 16);
      acc[t] = __builtin_amdgcn_wmma_f32_16x16x32_f16(
          false, a, false, b, (short)0, acc[t], false, false);
    }
  }
#pragma unroll
  for (int t = 0; t < 8; ++t) {
    float bv = bias[16 * t + ln];
#pragma unroll
    for (int r = 0; r < 8; ++r) {
      int ro = i0 + wave * 16 + grp * 8 + r;     // C layout: lanes<16 M=r, else M=r+8
      Xp[(size_t)ro * OUTF + 16 * t + ln] = acc[t][r] + bv;
    }
  }
}

// ------------- K2: a_dst[j] = X'[j] . S[128:], global max via atomicMax -------------
__global__ __launch_bounds__(256) void gat_k2_adst(
    const float* __restrict__ Xp, const float* __restrict__ S,
    float* __restrict__ a_dst, unsigned* __restrict__ maxcell) {
  int r = blockIdx.x * 256 + threadIdx.x;
  const float* xr = Xp + (size_t)r * OUTF;
  float s = 0.f;
#pragma unroll 4
  for (int c = 0; c < OUTF; ++c) s += xr[c] * S[OUTF + c];
  a_dst[r] = s;
  atomicMax(maxcell, enc_f32(s));
}

// ------------- K3: Yt[c][j] = e[j]*X'[j][c], Yt[128][j]=e[j], pad cols = 0 -------------
__global__ __launch_bounds__(256) void gat_k3_build(
    const float* __restrict__ Xp, const float* __restrict__ a_dst,
    const unsigned* __restrict__ maxcell, _Float16* __restrict__ Yt) {
  int j = blockIdx.x * 256 + threadIdx.x;
  float m = dec_f32(*maxcell);
  float e = expf(a_dst[j] - m);                  // <= 1, softmax shift cancels
  const float* xr = Xp + (size_t)j * OUTF;
#pragma unroll 4
  for (int c = 0; c < OUTF; ++c)
    Yt[(size_t)c * NNODES + j] = (_Float16)(e * xr[c]);
  Yt[(size_t)OUTF * NNODES + j] = (_Float16)e;   // denominator column
#pragma unroll
  for (int c = OUTF + 1; c < NCOLS; ++c)
    Yt[(size_t)c * NNODES + j] = (_Float16)0.f;  // zero padding
}

// ------------- K4: out = sigmoid( (mask @ Y)[:, :128] / (mask @ Y)[:,128] ) -------------
// block = 128 thr = 4 waves; each wave: 16 rows x 144 cols (9 f32 accumulators).
// Streams 256MB of adj exactly once; Yt tile staged into LDS (async DMA path if available).
// Self-loop (mask diagonal) is added in the epilogue, not in the inner loop.
__global__ __launch_bounds__(128) void gat_k4_aggregate(
    const int* __restrict__ adj, const _Float16* __restrict__ Yt,
    float* __restrict__ out) {
  __shared__ _Float16 Bs[NCOLS * LDSP];
  const int tid = threadIdx.x, wave = tid >> 5, lane = tid & 31;
  const int ln = lane & 15, grp = lane >> 4;
  const int i0 = blockIdx.x * 64;
  const int row = i0 + wave * 16 + ln;
  v8f zero = {};
  v8f acc[9];
#pragma unroll
  for (int t = 0; t < 9; ++t) acc[t] = zero;

  for (int k0 = 0; k0 < NNODES; k0 += 32) {
    // ---- stage B-tile: 144 cols x 32 K, as 576 8-half (16B) chunks ----
#if defined(USE_ASYNC_LDS)
    for (int c = tid; c < 576; c += 128) {
      int n = c >> 2, ko = (c & 3) << 3;
      __builtin_amdgcn_global_load_async_to_lds_b128(
          (AS1 v4i*)(v4i*)(Yt + (size_t)n * NNODES + k0 + ko),
          (AS3 v4i*)(v4i*)&Bs[n * LDSP + ko], 0, 0);
    }
#else
    for (int c = tid; c < 576; c += 128) {
      int n = c >> 2, ko = (c & 3) << 3;
      *(v8h*)&Bs[n * LDSP + ko] = *(const v8h*)(Yt + (size_t)n * NNODES + k0 + ko);
    }
#endif

    // ---- build A (0/1 mask) while the staging is in flight ----
    v16h a;
    const int colbase = k0 + grp * 8;
    const int* ap = adj + (size_t)row * NNODES + colbase;
    mask8(a, 0, ap);          // K base..base+7
    mask8(a, 8, ap + 16);     // K base+16..base+23
    if (k0 + 32 < NNODES) __builtin_prefetch(ap + 32, 0, 1);

#if defined(USE_ASYNC_LDS)
    __builtin_amdgcn_s_wait_asynccnt(0);
#endif
    __syncthreads();

#pragma unroll
    for (int t = 0; t < 9; ++t) {
      const _Float16* bp = &Bs[(16 * t + ln) * LDSP + grp * 8];
      v16h b; lds8(b, 0, bp); lds8(b, 8, bp + 16);
      acc[t] = __builtin_amdgcn_wmma_f32_16x16x32_f16(
          false, a, false, b, (short)0, acc[t], false, false);
    }
    __syncthreads();
  }

  // ---- epilogue: add the self-loop contribution e[ro]*X'[ro][col] (diag of mask) ----
  // Yt[col][ro] is exactly that value for cols 0..128, zero in the padding cols.
#pragma unroll
  for (int t = 0; t < 9; ++t)
#pragma unroll
    for (int r = 0; r < 8; ++r) {
      int ro = i0 + wave * 16 + grp * 8 + r;
      acc[t][r] += (float)Yt[(size_t)(16 * t + ln) * NNODES + ro];
    }

  // denominator column (N=128) lives in acc[8] at ln==0; broadcast within half-wave
  float den[8];
#pragma unroll
  for (int r = 0; r < 8; ++r) den[r] = __shfl(acc[8][r], lane & 16, 32);

#pragma unroll
  for (int t = 0; t < 8; ++t)
#pragma unroll
    for (int r = 0; r < 8; ++r) {
      int ro = i0 + wave * 16 + grp * 8 + r;
      float x = acc[t][r] / den[r];
      out[(size_t)ro * OUTF + 16 * t + ln] = 1.0f / (1.0f + expf(-x));
    }
}

extern "C" void kernel_launch(void* const* d_in, const int* in_sizes, int n_in,
                              void* d_out, int out_size, void* d_ws, size_t ws_size,
                              hipStream_t stream) {
  const float* X   = (const float*)d_in[0];   // (8192, 256)
  const int*   adj = (const int*)  d_in[1];   // (8192, 8192)
  const float* W   = (const float*)d_in[2];   // (128, 256)
  const float* b   = (const float*)d_in[3];   // (128,)
  const float* S   = (const float*)d_in[4];   // (256,)
  float* out = (float*)d_out;                 // (8192, 128) f32

  char* ws = (char*)d_ws;
  float*    Xp      = (float*)(ws);                                   // 4 MB
  float*    a_dst   = (float*)(ws + (size_t)4 * 1024 * 1024);         // 32 KB
  unsigned* maxcell = (unsigned*)(ws + (size_t)4 * 1024 * 1024 + 32 * 1024);
  _Float16* Yt      = (_Float16*)(ws + (size_t)4 * 1024 * 1024 + 32 * 1024 + 64); // 2.25 MB

  hipLaunchKernelGGL(gat_k0_init,      dim3(1),   dim3(1),   0, stream, maxcell);
  hipLaunchKernelGGL(gat_k1_linear,    dim3(64),  dim3(256), 0, stream, X, W, b, Xp);
  hipLaunchKernelGGL(gat_k2_adst,      dim3(32),  dim3(256), 0, stream, Xp, S, a_dst, maxcell);
  hipLaunchKernelGGL(gat_k3_build,     dim3(32),  dim3(256), 0, stream, Xp, a_dst, maxcell, Yt);
  hipLaunchKernelGGL(gat_k4_aggregate, dim3(128), dim3(128), 0, stream, adj, Yt, out);
}
